// MultiHeadAttention_67061619360129
// MI455X (gfx1250) — matrix operand order repaired and checked by reading, not compiled, not run on hardware
//
#include <hip/hip_runtime.h>

typedef unsigned short u16;
typedef unsigned int   u32;

typedef __attribute__((ext_vector_type(16))) __bf16 v16bf;
typedef __attribute__((ext_vector_type(8)))  float  v8f;
typedef __attribute__((ext_vector_type(4)))  u32    v4u;
typedef __attribute__((ext_vector_type(4)))  float  v4f;

#define WMMA_BF16(a,b,c) __builtin_amdgcn_wmma_f32_16x16x32_bf16(false,(a),false,(b),(short)0,(c),false,false)

#if __has_builtin(__builtin_amdgcn_global_load_async_to_lds_b128) && __has_builtin(__builtin_amdgcn_s_wait_asynccnt)
#define HAVE_ASYNC 1
typedef int v4i_gnu __attribute__((vector_size(16)));   // matches builtin's pointee type
#define GAS __attribute__((address_space(1)))
#define LAS __attribute__((address_space(3)))
#else
#define HAVE_ASYNC 0
#endif

// Problem constants
#define B_  4
#define T_  2048
#define D_  1024
#define H_  16
#define DH_ 64
#define M_  (B_*T_)      // 8192
#define N1_ (3*D_)       // 3072

union FragU { v4u u[2]; v16bf v; };

__device__ __forceinline__ u16 f32_to_bf16(float f) {
  u32 u = __float_as_uint(f);
  u32 r = 0x7FFFu + ((u >> 16) & 1u);
  return (u16)((u + r) >> 16);
}

// A-operand fragment: per-lane elems {e..e+7, e+16..e+23}; caller adds (lane>>4)*8
__device__ __forceinline__ v16bf ld_frag_a(const u16* p) {
  FragU f; f.u[0] = *(const v4u*)p; f.u[1] = *(const v4u*)(p + 16); return f.v;
}
// B-operand fragment: per-lane elems {e..e+7, e+16..e+23}; caller adds (lane>>4)*8
__device__ __forceinline__ v16bf ld_frag_b(const u16* p) {
  FragU f; f.u[0] = *(const v4u*)p; f.u[1] = *(const v4u*)(p + 16); return f.v;
}

__device__ __forceinline__ v8f zero8() {
  v8f z;
#pragma unroll
  for (int e = 0; e < 8; ++e) z[e] = 0.0f;
  return z;
}

__device__ __forceinline__ v16bf ones_frag() {   // bf16 1.0 splat
  FragU f;
  v4u o = { 0x3F803F80u, 0x3F803F80u, 0x3F803F80u, 0x3F803F80u };
  f.u[0] = o; f.u[1] = o;
  return f.v;
}

// 16-lane max reduction entirely in VALU via DPP (xor1, xor2, xor~4, xor~8)
#define ROWMAX_STEP(x, ctrl) do {                                              \
    int _xi = __float_as_int(x);                                               \
    int _yi = __builtin_amdgcn_update_dpp(_xi, _xi, (ctrl), 0xF, 0xF, false);  \
    (x) = fmaxf((x), __int_as_float(_yi));                                     \
  } while (0)
__device__ __forceinline__ float rowmax16(float x) {
  ROWMAX_STEP(x, 0xB1);   // quad_perm [1,0,3,2]  (xor 1)
  ROWMAX_STEP(x, 0x4E);   // quad_perm [2,3,0,1]  (xor 2)
  ROWMAX_STEP(x, 0x141);  // row_half_mirror      (pairs groups of 4)
  ROWMAX_STEP(x, 0x140);  // row_mirror           (pairs groups of 8)
  return x;
}

#if HAVE_ASYNC
__device__ __forceinline__ void async_cp16(u16* l, const u16* g) {
  __builtin_amdgcn_global_load_async_to_lds_b128(
      (GAS v4i_gnu*)(unsigned long long)(g),
      (LAS v4i_gnu*)(unsigned)(unsigned long long)(l),
      0, 0);
}
#endif

// ---------------- elementwise f32 -> bf16 ----------------
__global__ __launch_bounds__(256) void cvt_bf16_kernel(const float* __restrict__ src,
                                                       u16* __restrict__ dst, int n4) {
  int i = blockIdx.x * 256 + threadIdx.x;
  if (i >= n4) return;
  v4f v = ((const v4f*)src)[i];
  u32 lo = (u32)f32_to_bf16(v[0]) | ((u32)f32_to_bf16(v[1]) << 16);
  u32 hi = (u32)f32_to_bf16(v[2]) | ((u32)f32_to_bf16(v[3]) << 16);
  ((u32*)dst)[2*i]   = lo;
  ((u32*)dst)[2*i+1] = hi;
}

// -------- transpose + convert: W[K][N] f32 -> WT[N][K] bf16 --------
__global__ __launch_bounds__(256) void transpose_cvt_kernel(const float* __restrict__ W,
                                                            u16* __restrict__ WT,
                                                            int K, int N) {
  __shared__ float tile[32][33];
  int k0 = blockIdx.y * 32, n0 = blockIdx.x * 32;
  int tx = threadIdx.x & 31, ty = threadIdx.x >> 5;
#pragma unroll
  for (int i = 0; i < 32; i += 8)
    tile[ty + i][tx] = W[(size_t)(k0 + ty + i) * N + n0 + tx];
  __syncthreads();
#pragma unroll
  for (int i = 0; i < 32; i += 8)
    WT[(size_t)(n0 + ty + i) * K + k0 + tx] = f32_to_bf16(tile[tx][ty + i]);
}

// ---------------- tiled bf16 GEMM via WMMA ----------------
// C[M][N] = A[M][K] * Bt[N][K]^T (+bias).  MODE 0: qkv scatter; MODE 1: f32 out.
template<int MODE>
__global__ __launch_bounds__(256) void gemm_bf16_kernel(
    const u16* __restrict__ A, const u16* __restrict__ Bt,
    const float* __restrict__ bias,
    u16* __restrict__ qb, u16* __restrict__ kb, u16* __restrict__ vtb,
    float* __restrict__ outF, int Kdim, int Ncols)
{
  constexpr int LDSS = 40;                      // padded row stride (conflict-free)
  __shared__ u16 lA[2][128 * LDSS];
  __shared__ u16 lB[2][128 * LDSS];

  const int tid  = threadIdx.x;
  const int lane = tid & 31;
  const int w    = tid >> 5;
  const int l15  = lane & 15;
  const int ah   = (lane >> 4) << 3;
  const int bh   = (lane >> 4) << 3;
  const int m_blk = blockIdx.y << 7;
  const int n_blk = blockIdx.x << 7;
  const int wm = (w >> 2) << 6;                 // 0/64
  const int wn = (w & 3) << 5;                  // 0/32/64/96

  const int srow = tid >> 2;                    // 0..63
  const int sseg = (tid & 3) << 3;              // 0/8/16/24 elems
  const u16* gA = A  + (size_t)(m_blk + srow) * Kdim;
  const u16* gB = Bt + (size_t)(n_blk + srow) * Kdim;
  const size_t rstep = (size_t)64 * Kdim;

  v8f acc[4][2];
#pragma unroll
  for (int i = 0; i < 4; ++i)
#pragma unroll
    for (int j = 0; j < 2; ++j) acc[i][j] = zero8();

  const int nk = Kdim >> 5;

#if HAVE_ASYNC
  { // issue K-tile 0 straight into LDS via async copy unit
    async_cp16(&lA[0][srow * LDSS + sseg],        gA + sseg);
    async_cp16(&lA[0][(srow + 64) * LDSS + sseg], gA + rstep + sseg);
    async_cp16(&lB[0][srow * LDSS + sseg],        gB + sseg);
    async_cp16(&lB[0][(srow + 64) * LDSS + sseg], gB + rstep + sseg);
  }
#else
  {
    v4u a0 = *(const v4u*)(gA + sseg);
    v4u a1 = *(const v4u*)(gA + rstep + sseg);
    v4u b0 = *(const v4u*)(gB + sseg);
    v4u b1 = *(const v4u*)(gB + rstep + sseg);
    *(v4u*)&lA[0][srow * LDSS + sseg]        = a0;
    *(v4u*)&lA[0][(srow + 64) * LDSS + sseg] = a1;
    *(v4u*)&lB[0][srow * LDSS + sseg]        = b0;
    *(v4u*)&lB[0][(srow + 64) * LDSS + sseg] = b1;
  }
#endif

#if !HAVE_ASYNC
  v4u pa0, pa1, pb0, pb1;
#endif
  for (int kt = 0; kt < nk; ++kt) {
    const int cur = kt & 1;
    const int nxt = cur ^ 1;
    if (kt + 1 < nk) {
      const int ko = (kt + 1) << 5;
#if HAVE_ASYNC
      async_cp16(&lA[nxt][srow * LDSS + sseg],        gA + ko + sseg);
      async_cp16(&lA[nxt][(srow + 64) * LDSS + sseg], gA + rstep + ko + sseg);
      async_cp16(&lB[nxt][srow * LDSS + sseg],        gB + ko + sseg);
      async_cp16(&lB[nxt][(srow + 64) * LDSS + sseg], gB + rstep + ko + sseg);
#else
      pa0 = *(const v4u*)(gA + ko + sseg);
      pa1 = *(const v4u*)(gA + rstep + ko + sseg);
      pb0 = *(const v4u*)(gB + ko + sseg);
      pb1 = *(const v4u*)(gB + rstep + ko + sseg);
#endif
    }
#if HAVE_ASYNC
    // in-order completion: waiting <=4 guarantees tile kt's 4 copies landed
    if (kt + 1 < nk) __builtin_amdgcn_s_wait_asynccnt(4);
    else             __builtin_amdgcn_s_wait_asynccnt(0);
#endif
    __syncthreads();

    const u16* la = lA[cur];
    const u16* lb = lB[cur];
    v16bf af[4], bfr[2];
#pragma unroll
    for (int mi = 0; mi < 4; ++mi)
      af[mi] = ld_frag_a(la + (wm + mi * 16 + l15) * LDSS + ah);
#pragma unroll
    for (int ni = 0; ni < 2; ++ni)
      bfr[ni] = ld_frag_b(lb + (wn + ni * 16 + l15) * LDSS + bh);
#pragma unroll
    for (int mi = 0; mi < 4; ++mi)
#pragma unroll
      for (int ni = 0; ni < 2; ++ni)
        acc[mi][ni] = WMMA_BF16(af[mi], bfr[ni], acc[mi][ni]);
    __syncthreads();
#if !HAVE_ASYNC
    if (kt + 1 < nk) {
      *(v4u*)&lA[nxt][srow * LDSS + sseg]        = pa0;
      *(v4u*)&lA[nxt][(srow + 64) * LDSS + sseg] = pa1;
      *(v4u*)&lB[nxt][srow * LDSS + sseg]        = pb0;
      *(v4u*)&lB[nxt][(srow + 64) * LDSS + sseg] = pb1;
    }
#endif
  }

  // epilogue
#pragma unroll
  for (int mi = 0; mi < 4; ++mi)
#pragma unroll
    for (int ni = 0; ni < 2; ++ni) {
      const int n = n_blk + wn + ni * 16 + l15;
      const float bv = bias[n];
      const int mbase = m_blk + wm + mi * 16 + ah;
#pragma unroll
      for (int r = 0; r < 8; ++r) {
        const int m = mbase + r;
        const float val = acc[mi][ni][r] + bv;
        if (MODE == 0) {
          const int bi  = m >> 11, t = m & 2047;
          const int sel = n >> 10, hh = (n >> 6) & 15, d = n & 63;
          const int bhh = (bi << 4) + hh;
          const u16 bw  = f32_to_bf16(val);
          const size_t qk = ((size_t)bhh * T_ + t) * DH_ + d;
          if (sel == 0)      qb[qk] = bw;
          else if (sel == 1) kb[qk] = bw;
          else               vtb[((size_t)bhh * DH_ + d) * T_ + t] = bw;
        } else {
          outF[(size_t)m * Ncols + n] = val;
        }
      }
    }
}

// ---------------- flash-style attention, WMMA bf16 ----------------
__global__ __launch_bounds__(256) void attn_kernel(
    const u16* __restrict__ qb, const u16* __restrict__ kb,
    const u16* __restrict__ vtb, const int* __restrict__ mask,
    u16* __restrict__ attn)
{
  __shared__ u16 pl[8][16 * 40];                // wave-private P tiles (16x32, stride 40)
  const int tid  = threadIdx.x;
  const int lane = tid & 31;
  const int w    = tid >> 5;
  const int l15  = lane & 15;
  const int ah   = (lane >> 4) << 3;            // A half / C row-half offset
  const int bhf  = (lane >> 4) << 3;            // B half offset
  const int b    = blockIdx.z, h = blockIdx.y;
  const int bhid = b * H_ + h;
  const int m0   = blockIdx.x * 128 + w * 16;   // query-tile base (t index)

  // resident q A-fragments (16 x 64)
  const u16* qrow = qb + ((size_t)bhid * T_ + m0 + l15) * DH_;
  v16bf aq0 = ld_frag_a(qrow + ah);
  v16bf aq1 = ld_frag_a(qrow + 32 + ah);
  const v16bf onesb = ones_frag();

  v8f o[4];
#pragma unroll
  for (int t4 = 0; t4 < 4; ++t4) o[t4] = zero8();
  v8f ol = zero8();                             // row-sum accumulator (P @ ones)
  float mr[8];
#pragma unroll
  for (int r = 0; r < 8; ++r) mr[r] = -1e30f;

  const int* mrow = mask + b * T_;
  u16* pw = pl[w];

  for (int jj = 0; jj < T_ / 32; ++jj) {
    const int tk = jj * 32;
    const u16* k0r = kb + ((size_t)bhid * T_ + tk + l15) * DH_;
    const u16* k1r = k0r + 16 * DH_;
    v16bf bk00 = ld_frag_b(k0r + bhf),      bk01 = ld_frag_b(k0r + 32 + bhf);
    v16bf bk10 = ld_frag_b(k1r + bhf),      bk11 = ld_frag_b(k1r + 32 + bhf);

    v8f s0 = WMMA_BF16(aq0, bk00, zero8());
    s0     = WMMA_BF16(aq1, bk01, s0);
    v8f s1 = WMMA_BF16(aq0, bk10, zero8());
    s1     = WMMA_BF16(aq1, bk11, s1);

    const int mv0 = mrow[tk + l15];
    const int mv1 = mrow[tk + 16 + l15];

#pragma unroll
    for (int r = 0; r < 8; ++r) {
      float x0 = mv0 ? s0[r] * 0.125f : -1000.0f;
      float x1 = mv1 ? s1[r] * 0.125f : -1000.0f;
      float tm = rowmax16(fmaxf(x0, x1));       // 16-lane DPP max, no LDS
      const float mn   = fmaxf(mr[r], tm);
      const float corr = __expf(mr[r] - mn);
      mr[r] = mn;
      const float p0 = __expf(x0 - mn);
      const float p1 = __expf(x1 - mn);
#pragma unroll
      for (int t4 = 0; t4 < 4; ++t4) o[t4][r] *= corr;
      ol[r] *= corr;
      const int prow = (r + ah) * 40;
      pw[prow + l15]      = f32_to_bf16(p0);
      pw[prow + 16 + l15] = f32_to_bf16(p1);
    }

    v16bf ap = ld_frag_a(pw + l15 * 40 + ah);   // P as A-operand (16 x 32)
    ol = WMMA_BF16(ap, onesb, ol);              // row sums ride the matrix pipe
#pragma unroll
    for (int t4 = 0; t4 < 4; ++t4) {
      const u16* vr = vtb + ((size_t)bhid * DH_ + t4 * 16 + l15) * T_ + tk;
      v16bf bv = ld_frag_b(vr + bhf);
      o[t4] = WMMA_BF16(ap, bv, o[t4]);
    }
  }

  float inv[8];
#pragma unroll
  for (int r = 0; r < 8; ++r) inv[r] = 1.0f / ol[r];
#pragma unroll
  for (int t4 = 0; t4 < 4; ++t4)
#pragma unroll
    for (int r = 0; r < 8; ++r) {
      const int trow = m0 + r + ah;
      const int col  = h * DH_ + t4 * 16 + l15;
      attn[((size_t)b * T_ + trow) * D_ + col] = f32_to_bf16(o[t4][r] * inv[r]);
    }
}

// ---------------- host-side orchestration ----------------
extern "C" void kernel_launch(void* const* d_in, const int* in_sizes, int n_in,
                              void* d_out, int out_size, void* d_ws, size_t ws_size,
                              hipStream_t stream) {
  const float* x     = (const float*)d_in[0];
  const int*   mask  = (const int*)  d_in[1];
  const float* Wqkv  = (const float*)d_in[2];
  const float* bqkv  = (const float*)d_in[3];
  const float* Wout  = (const float*)d_in[4];
  const float* bout  = (const float*)d_in[5];
  float*       out   = (float*)d_out;
  char*        ws    = (char*)d_ws;

  u16* xb    = (u16*)(ws);                         // 16 MiB  x  bf16 [8192][1024]
  u16* wqkvT = (u16*)(ws + 16777216);              //  6 MiB  WqkvT [3072][1024]
  u16* woutT = (u16*)(ws + 23068672);              //  2 MiB  WoutT [1024][1024]
  u16* qb    = (u16*)(ws + 25165824);              // 16 MiB  q  [B,H,T,64]
  u16* kb    = (u16*)(ws + 41943040);              // 16 MiB  k  [B,H,T,64]
  u16* vtb   = (u16*)(ws + 58720256);              // 16 MiB  vT [B,H,64,T]
  u16* attn  = (u16*)(ws + 75497472);              // 16 MiB  attn bf16 [8192][1024]

  cvt_bf16_kernel<<<(M_ * D_ / 4 + 255) / 256, 256, 0, stream>>>(x, xb, M_ * D_ / 4);
  transpose_cvt_kernel<<<dim3(N1_ / 32, D_ / 32), 256, 0, stream>>>(Wqkv, wqkvT, D_, N1_);
  transpose_cvt_kernel<<<dim3(D_ / 32,  D_ / 32), 256, 0, stream>>>(Wout, woutT, D_, D_);

  gemm_bf16_kernel<0><<<dim3(N1_ / 128, M_ / 128), 256, 0, stream>>>(
      xb, wqkvT, bqkv, qb, kb, vtb, nullptr, D_, N1_);

  attn_kernel<<<dim3(T_ / 128, H_, B_), 256, 0, stream>>>(qb, kb, vtb, mask, attn);

  gemm_bf16_kernel<1><<<dim3(D_ / 128, M_ / 128), 256, 0, stream>>>(
      attn, woutT, bout, nullptr, nullptr, nullptr, out, D_, D_);
}
